// MockMoELayer_38225208934445
// MI455X (gfx1250) — compile-verified, matrix operand order checked by reading, not run on hardware
//
#include <hip/hip_runtime.h>
#include <hip/hip_bf16.h>

// ---------------------------------------------------------------------------
// MoE (top-2 of 8 experts + shared expert), bf16 WMMA pipeline for gfx1250.
//
// Pipeline (all on `stream`, deterministic):
//   1. zero_init: out = 0, bucket counters = 0
//   2. conv_bf16 x7: f32 weights AND activations -> bf16 copies in workspace
//      (halves HBM traffic, feeds v_wmma_f32_16x16x32_bf16 directly)
//   3. router: logits -> softmax -> top2 -> renorm; append (token, weight) into
//      per-(slot, expert) buckets (integer atomics only; per-token math is
//      independent of bucket position -> deterministic output)
//   4. moe_mlp x3 (shared, slot0, slot1): async-DMA-gather 64 token rows into
//      LDS (global_load_async_to_lds_b128, ASYNCcnt), fused
//      gate/up GEMM -> silu*up*router_weight -> hmid (LDS, bf16) -> down GEMM,
//      chunked over I; accumulate `out +=` (each token owned by one block per
//      launch -> race-free, fixed summation order across launches).
// ---------------------------------------------------------------------------

typedef float   v8f   __attribute__((ext_vector_type(8)));
typedef __bf16  v8bf  __attribute__((ext_vector_type(8)));
typedef __bf16  v16bf __attribute__((ext_vector_type(16)));
typedef int     v4i   __attribute__((ext_vector_type(4)));

#define HIDDEN 2048
#define INTER  1408
#define NEXP   8
#define TOKENS 2048          // BATCH * SEQ
#define TM     64            // tokens per workgroup tile
#define XP     2056          // LDS pitch (halves) for X tile (2048 + 8 pad -> bank spread)
#define HP     360           // LDS pitch (halves) for hmid chunk (352 + 8 pad)
#define IC     352           // I-chunk (1408 / 4), multiple of 32
#define NCHUNK 4
#define EPW    (INTER * HIDDEN)   // 2883584 elems: per-expert matrix size (all 3 matrices)

#define AS1 __attribute__((address_space(1)))
#define AS3 __attribute__((address_space(3)))

__device__ __forceinline__ unsigned short f2bf(float f) {
    union { float f; unsigned u; } v; v.f = f;
    unsigned r = v.u + 0x7FFFu + ((v.u >> 16) & 1u);   // round-to-nearest-even
    return (unsigned short)(r >> 16);
}

__device__ __forceinline__ v16bf cat8(v8bf lo, v8bf hi) {
    return __builtin_shufflevector(lo, hi, 0, 1, 2, 3, 4, 5, 6, 7,
                                           8, 9, 10, 11, 12, 13, 14, 15);
}

// Async 16-byte global -> LDS copy (tracked with ASYNCcnt, no VGPR round-trip).
__device__ __forceinline__ void async_g2l_b128(const unsigned short* g,
                                               unsigned short* l) {
#if __has_builtin(__builtin_amdgcn_global_load_async_to_lds_b128)
    __builtin_amdgcn_global_load_async_to_lds_b128(
        (AS1 v4i*)(g), (AS3 v4i*)(l), 0, 0);
#else
    unsigned ldsoff = (unsigned)(unsigned long long)(AS3 void*)(l);
    asm volatile("global_load_async_to_lds_b128 %0, %1, off"
                 :: "v"(ldsoff), "v"((unsigned long long)g) : "memory");
#endif
}

__device__ __forceinline__ void wait_async0() {
#if __has_builtin(__builtin_amdgcn_s_wait_asynccnt)
    __builtin_amdgcn_s_wait_asynccnt(0);
#else
    asm volatile("s_wait_asynccnt 0x0" ::: "memory");
#endif
}

// ---------------------------------------------------------------------------
// out = 0, bucket counters = 0
__global__ __launch_bounds__(256) void zero_init_kernel(float4* __restrict__ out4,
                                                        unsigned* __restrict__ cnt) {
    int i = blockIdx.x * 256 + threadIdx.x;            // grid sized exactly
    float4 z; z.x = 0.f; z.y = 0.f; z.z = 0.f; z.w = 0.f;
    out4[i] = z;
    if (blockIdx.x == 0 && threadIdx.x < 16) cnt[threadIdx.x] = 0u;
}

// ---------------------------------------------------------------------------
// f32 -> bf16 conversion (4 elems/thread/iter, packed 8B stores)
__global__ __launch_bounds__(256) void conv_bf16_kernel(const float4* __restrict__ src,
                                                        unsigned long long* __restrict__ dst,
                                                        int n4) {
    for (int i = blockIdx.x * blockDim.x + threadIdx.x; i < n4;
         i += gridDim.x * blockDim.x) {
        float4 v = src[i];
        dst[i] = (unsigned long long)f2bf(v.x)
               | ((unsigned long long)f2bf(v.y) << 16)
               | ((unsigned long long)f2bf(v.z) << 32)
               | ((unsigned long long)f2bf(v.w) << 48);
    }
}

// ---------------------------------------------------------------------------
// Router: one wave32 per token. softmax over 8 experts, top-2, renormalize,
// append token into per-(slot, expert) bucket. Runs on f32 activations so
// top-k decisions match the reference.
__global__ __launch_bounds__(256) void router_kernel(const float* __restrict__ x,
                                                     const float* __restrict__ rw,
                                                     unsigned* __restrict__ cnt,
                                                     int* __restrict__ bucket,
                                                     float* __restrict__ wbuf) {
    const int lane = threadIdx.x & 31;
    const int wave = threadIdx.x >> 5;
    const int t = blockIdx.x * 8 + wave;

    float acc[NEXP];
#pragma unroll
    for (int e = 0; e < NEXP; ++e) acc[e] = 0.0f;

    const float* xr = x + (size_t)t * HIDDEN;
    for (int h = lane; h < HIDDEN; h += 32) {
        float xv = xr[h];
#pragma unroll
        for (int e = 0; e < NEXP; ++e) acc[e] += xv * rw[e * HIDDEN + h];
    }
#pragma unroll
    for (int off = 16; off > 0; off >>= 1) {
#pragma unroll
        for (int e = 0; e < NEXP; ++e) acc[e] += __shfl_xor(acc[e], off, 32);
    }

    if (lane == 0) {
        float m = acc[0];
#pragma unroll
        for (int e = 1; e < NEXP; ++e) m = fmaxf(m, acc[e]);
        float p[NEXP];
#pragma unroll
        for (int e = 0; e < NEXP; ++e) p[e] = __expf(acc[e] - m);
        int i0 = 0;
#pragma unroll
        for (int e = 1; e < NEXP; ++e) if (p[e] > p[i0]) i0 = e;
        int i1 = (i0 == 0) ? 1 : 0;
#pragma unroll
        for (int e = 0; e < NEXP; ++e) if (e != i0 && p[e] > p[i1]) i1 = e;
        float w0 = p[i0] / (p[i0] + p[i1]);   // softmax denom cancels in renorm
        float w1 = 1.0f - w0;

        unsigned pos0 = atomicAdd(&cnt[0 * NEXP + i0], 1u);
        bucket[(0 * NEXP + i0) * TOKENS + pos0] = t;
        wbuf  [(0 * NEXP + i0) * TOKENS + pos0] = w0;
        unsigned pos1 = atomicAdd(&cnt[1 * NEXP + i1], 1u);
        bucket[(1 * NEXP + i1) * TOKENS + pos1] = t;
        wbuf  [(1 * NEXP + i1) * TOKENS + pos1] = w1;
    }
}

// ---------------------------------------------------------------------------
// Fused expert MLP over a gathered 64-token tile.
//   slot < 0  : shared expert (identity token list, weight 1.0)
//   slot 0/1  : routed experts, token list from bucket[slot][e]
// LDS: Xs 64x2056 bf16 (257 KB) + Hm 64x360 bf16 (45 KB) + token ids/weights.
__global__ __launch_bounds__(256) void moe_mlp_kernel(const unsigned short* __restrict__ xbf,
                                                      const unsigned short* __restrict__ gw,
                                                      const unsigned short* __restrict__ uw,
                                                      const unsigned short* __restrict__ dw,
                                                      const unsigned* __restrict__ cnt,
                                                      const int* __restrict__ bucket,
                                                      const float* __restrict__ wbuf,
                                                      float* __restrict__ out,
                                                      int slot) {
    extern __shared__ char smem[];
    unsigned short* Xs = (unsigned short*)smem;          // TM x XP halves
    unsigned short* Hm = Xs + (size_t)TM * XP;           // TM x HP halves
    int*   toksS = (int*)(Hm + (size_t)TM * HP);
    float* twS   = (float*)(toksS + TM);

    const int tid  = threadIdx.x;
    const int wave = tid >> 5;
    const int lane = tid & 31;
    const int nl   = lane & 15;        // row (A) / col (B,C) index within 16-tile
    const int ko   = (lane >> 4) * 8;  // K half-select per ISA 16-bit layouts

    const int e    = blockIdx.x;
    const int tile = blockIdx.y;

    int nt;
    if (slot < 0) {
        nt = TM;
        if (tid < TM) { toksS[tid] = tile * TM + tid; twS[tid] = 1.0f; }
    } else {
        const int bidx = slot * NEXP + e;
        const int cn   = (int)cnt[bidx];
        const int base = tile * TM;
        if (base >= cn) return;                 // uniform across block
        nt = cn - base; if (nt > TM) nt = TM;
        if (tid < TM) {
            int j = (tid < nt) ? tid : 0;       // pad rows alias row 0, weight 0
            toksS[tid] = bucket[bidx * TOKENS + base + j];
            twS[tid]   = (tid < nt) ? wbuf[bidx * TOKENS + base + tid] : 0.0f;
        }
    }
    __syncthreads();

    // Async DMA-gather of the 64 x 2048 bf16 activation tile into LDS:
    // 16 B per issue, tracked with ASYNCcnt (no VGPR round-trip, no cvt VALU).
    for (int i = tid; i < TM * (HIDDEN / 8); i += 256) {
        int r = i >> 8;                         // HIDDEN/8 == 256
        int c = (i & 255) * 8;
        async_g2l_b128(xbf + (size_t)toksS[r] * HIDDEN + c,
                       Xs + (size_t)r * XP + c);
    }
    wait_async0();
    __syncthreads();

    const unsigned short* gwe = gw + (size_t)e * EPW;
    const unsigned short* uwe = uw + (size_t)e * EPW;
    const unsigned short* dwe = dw + (size_t)e * EPW;

    for (int ch = 0; ch < NCHUNK; ++ch) {
        const int i0c = ch * IC;

        // --- gate/up GEMMs + SwiGLU epilogue into Hm (chunk of I) ---
        for (int job = wave; job < 4 * (IC / 16); job += 8) {
            const int mt  = job & 3;
            const int nti = job >> 2;
            const int m0  = mt * 16;
            const int nglob = i0c + nti * 16 + nl;            // I row of weights
            const unsigned short* arow = Xs + (size_t)(m0 + nl) * XP;
            const unsigned short* grow = gwe + (size_t)nglob * HIDDEN;
            const unsigned short* urow = uwe + (size_t)nglob * HIDDEN;
            v8f accg = {0, 0, 0, 0, 0, 0, 0, 0};
            v8f accu = {0, 0, 0, 0, 0, 0, 0, 0};
#pragma unroll 4
            for (int k0 = 0; k0 < HIDDEN; k0 += 32) {
                if ((k0 & 255) == 0) {
                    __builtin_prefetch(grow + k0 + 512);      // global_prefetch_b8
                    __builtin_prefetch(urow + k0 + 512);
                }
                v16bf a  = cat8(*(const v8bf*)(arow + k0 + ko),
                                *(const v8bf*)(arow + k0 + 16 + ko));
                v16bf bg = cat8(*(const v8bf*)(grow + k0 + ko),
                                *(const v8bf*)(grow + k0 + 16 + ko));
                v16bf bu = cat8(*(const v8bf*)(urow + k0 + ko),
                                *(const v8bf*)(urow + k0 + 16 + ko));
                accg = __builtin_amdgcn_wmma_f32_16x16x32_bf16(
                           false, a, false, bg, (short)0, accg, false, false);
                accu = __builtin_amdgcn_wmma_f32_16x16x32_bf16(
                           false, a, false, bu, (short)0, accu, false, false);
            }
#pragma unroll
            for (int v = 0; v < 8; ++v) {
                int m = m0 + ko + v;            // C layout: M = (lane>>4)*8 + v
                float g = accg[v], u = accu[v];
                float hval = (g / (1.0f + __expf(-g))) * u * twS[m];
                Hm[(size_t)m * HP + nti * 16 + nl] = f2bf(hval);
            }
        }
        __syncthreads();

        // --- down GEMM: out += Hm(chunk) @ down_w[:, chunk]^T ---
        for (int job = wave; job < 4 * (HIDDEN / 16); job += 8) {
            const int mt = job & 3;
            const int ht = job >> 2;
            const int m0 = mt * 16;
            const int n0 = ht * 16;
            const unsigned short* arow = Hm  + (size_t)(m0 + nl) * HP;
            const unsigned short* drow = dwe + (size_t)(n0 + nl) * INTER + i0c;
            v8f acc = {0, 0, 0, 0, 0, 0, 0, 0};
#pragma unroll
            for (int kk = 0; kk < IC; kk += 32) {
                v16bf a = cat8(*(const v8bf*)(arow + kk + ko),
                               *(const v8bf*)(arow + kk + 16 + ko));
                v16bf b = cat8(*(const v8bf*)(drow + kk + ko),
                               *(const v8bf*)(drow + kk + 16 + ko));
                acc = __builtin_amdgcn_wmma_f32_16x16x32_bf16(
                          false, a, false, b, (short)0, acc, false, false);
            }
#pragma unroll
            for (int v = 0; v < 8; ++v) {
                int m = m0 + ko + v;
                if (m < nt)
                    out[(size_t)toksS[m] * HIDDEN + n0 + nl] += acc[v];
            }
        }
        __syncthreads();
    }
}

// ---------------------------------------------------------------------------
extern "C" void kernel_launch(void* const* d_in, const int* in_sizes, int n_in,
                              void* d_out, int out_size, void* d_ws, size_t ws_size,
                              hipStream_t stream) {
    (void)in_sizes; (void)n_in; (void)out_size; (void)ws_size;

    const float* x  = (const float*)d_in[0];
    const float* rw = (const float*)d_in[1];
    const float* gw = (const float*)d_in[2];
    const float* uw = (const float*)d_in[3];
    const float* dw = (const float*)d_in[4];
    const float* sg = (const float*)d_in[5];
    const float* su = (const float*)d_in[6];
    const float* sd = (const float*)d_in[7];
    float* out = (float*)d_out;

    // Workspace layout (all 256B-aligned):
    //   [0,256)        : 16 bucket counters (2 slots x 8 experts)
    //   [256, +128KB)  : bucket token ids
    //   [.., +128KB)   : bucket weights
    //   then bf16 copies: gate(8E), up(8E), down(8E), sgate, sup, sdown, x
    char* ws = (char*)d_ws;
    unsigned* cnt    = (unsigned*)ws;
    int*      bucket = (int*)(ws + 256);
    float*    wbuf   = (float*)(ws + 256 + 16 * TOKENS * 4);
    unsigned short* gbf = (unsigned short*)(ws + 256 + 2 * 16 * TOKENS * 4);
    unsigned short* ubf = gbf + (size_t)NEXP * EPW;
    unsigned short* dbf = ubf + (size_t)NEXP * EPW;
    unsigned short* sgb = dbf + (size_t)NEXP * EPW;
    unsigned short* sub = sgb + (size_t)EPW;
    unsigned short* sdb = sub + (size_t)EPW;
    unsigned short* xbf = sdb + (size_t)EPW;

    zero_init_kernel<<<TOKENS * HIDDEN / 4 / 256, 256, 0, stream>>>((float4*)out, cnt);

    const int n4e = NEXP * EPW / 4;
    const int n4s = EPW / 4;
    const int n4x = TOKENS * HIDDEN / 4;
    conv_bf16_kernel<<<2048, 256, 0, stream>>>((const float4*)gw, (unsigned long long*)gbf, n4e);
    conv_bf16_kernel<<<2048, 256, 0, stream>>>((const float4*)uw, (unsigned long long*)ubf, n4e);
    conv_bf16_kernel<<<2048, 256, 0, stream>>>((const float4*)dw, (unsigned long long*)dbf, n4e);
    conv_bf16_kernel<<<512,  256, 0, stream>>>((const float4*)sg, (unsigned long long*)sgb, n4s);
    conv_bf16_kernel<<<512,  256, 0, stream>>>((const float4*)su, (unsigned long long*)sub, n4s);
    conv_bf16_kernel<<<512,  256, 0, stream>>>((const float4*)sd, (unsigned long long*)sdb, n4s);
    conv_bf16_kernel<<<1024, 256, 0, stream>>>((const float4*)x,  (unsigned long long*)xbf, n4x);

    router_kernel<<<TOKENS / 8, 256, 0, stream>>>(x, rw, cnt, bucket, wbuf);

    const size_t shmem = (size_t)TM * XP * 2 + (size_t)TM * HP * 2 + TM * 4 + TM * 4;
    // shared expert: init-accumulate (out already zeroed)
    moe_mlp_kernel<<<dim3(1, TOKENS / TM), 256, shmem, stream>>>(
        xbf, sgb, sub, sdb, cnt, bucket, wbuf, out, -1);
    // routed experts, slot 0 then slot 1 (each token touched once per launch)
    moe_mlp_kernel<<<dim3(NEXP, TOKENS / TM), 256, shmem, stream>>>(
        xbf, gbf, ubf, dbf, cnt, bucket, wbuf, out, 0);
    moe_mlp_kernel<<<dim3(NEXP, TOKENS / TM), 256, shmem, stream>>>(
        xbf, gbf, ubf, dbf, cnt, bucket, wbuf, out, 1);
}
